// Net_1846835938183
// MI455X (gfx1250) — matrix operand order repaired and checked by reading, not compiled, not run on hardware
//
#include <hip/hip_runtime.h>
#include <hip/hip_bf16.h>

typedef __attribute__((ext_vector_type(16))) _Float16 v16h;
typedef __attribute__((ext_vector_type(8)))  float    v8f;

#define HEADS 8
#define NEG_SLOPE 0.2f

// ---------------------------------------------------------------------------
// Pack W[K,Nc] (f32, row-major) into WMMA B-fragment order, f16, zero-padded
// to 64 columns. One thread per (ktile, ntile, lane); each writes its lane's
// 16 f16 values contiguously so the GEMM can load them as one v16h.
// Layout: Wp[((kt*4 + nt)*32 + lane)*16 + i]
//   col = lane&15, kh = lane>>4, element i => B[kt*32 + kh*16 + i][nt*16+col]
// ---------------------------------------------------------------------------
__global__ void pack_w_kernel(const float* __restrict__ W, _Float16* __restrict__ Wp,
                              int K, int Nc) {
  int t = blockIdx.x * blockDim.x + threadIdx.x;
  int total = (K >> 5) * 4 * 32;
  if (t >= total) return;
  int lane = t & 31;
  int nt = (t >> 5) & 3;
  int kt = t >> 7;
  int col = lane & 15;
  int kh = lane >> 4;
  int n = nt * 16 + col;
  _Float16* dst = Wp + (size_t)t * 16;
  #pragma unroll
  for (int i = 0; i < 16; ++i) {
    float v = (n < Nc) ? W[(size_t)(kt * 32 + kh * 16 + i) * Nc + n] : 0.0f;
    dst[i] = (_Float16)v;
  }
}

// ---------------------------------------------------------------------------
// WMMA GEMM: D[M,Nc] = A[M,K] @ B[K,Nc]; fp32 A/D, prepacked f16 B fragments.
// Block = 128 threads = 4 waves; each wave computes a 16x16 tile, block covers
// 16 rows x 64 cols. Requires M % 16 == 0, K % 32 == 0, Nc <= 64.
// Inner loop: 4x b128 (A) + 2x b128 (B) + cvt + one v_wmma. No branches.
// ---------------------------------------------------------------------------
__global__ __launch_bounds__(128)
void gemm_f16_wmma(const float* __restrict__ A, const v16h* __restrict__ Bp,
                   float* __restrict__ D, int M, int K, int Nc) {
  const int wave = threadIdx.x >> 5;
  const int lane = threadIdx.x & 31;
  const int l16  = lane & 15;   // row (A) / column (D) selector
  const int kh   = lane >> 4;   // half-wave
  const int mbase = blockIdx.x << 4;
  const int ncol  = (wave << 4) + l16;

  const float* arow = A + (size_t)(mbase + l16) * K;
  const v16h* bfrag = Bp + (size_t)wave * 32 + lane;
  const int nKt = K >> 5;

  v8f acc = {};
  for (int kt = 0; kt < nKt; ++kt) {
    const float* ap = arow + kt * 32;
    v16h a;
    #pragma unroll
    for (int i = 0; i < 8; ++i) a[i]     = (_Float16)ap[kh * 16 + i];
    #pragma unroll
    for (int i = 0; i < 8; ++i) a[8 + i] = (_Float16)ap[kh * 16 + 8 + i];
    v16h b = bfrag[(size_t)kt * 128];   // (kt*4 + wave)*32 + lane
    acc = __builtin_amdgcn_wmma_f32_16x16x32_f16(false, a, false, b,
                                                 (short)0, acc, false, false);
  }
  if (ncol < Nc) {
    #pragma unroll
    for (int r = 0; r < 8; ++r)
      D[(size_t)(mbase + r + 8 * kh) * Nc + ncol] = acc[r];
  }
}

// Order-preserving float<->uint mapping for atomic segment-max.
__device__ __forceinline__ unsigned int f2key(float f) {
  unsigned int b = __float_as_uint(f);
  return (b & 0x80000000u) ? ~b : (b | 0x80000000u);
}
__device__ __forceinline__ float key2f(unsigned int k) {
  return __uint_as_float((k & 0x80000000u) ? (k & 0x7fffffffu) : ~k);
}

// ---------------------------------------------------------------------------
// Pass 1: per (edge, head): alpha = leaky_relu((a_l.x_j + a_r.x_i) *
// sigmoid(<x_i,x_j>)); store alpha; atomic segment-max over dst.
// Edges e >= E are self-loops (src = dst = e - E). C templated for unroll.
// ---------------------------------------------------------------------------
template <int C>
__global__ void edge_alpha_kernel(const float* __restrict__ H,
                                  const int* __restrict__ ei,
                                  const float* __restrict__ attL,
                                  const float* __restrict__ attR,
                                  float* __restrict__ alpha,
                                  unsigned int* __restrict__ nmax,
                                  int E, int Nn) {
  long long tid = (long long)blockIdx.x * blockDim.x + threadIdx.x;
  long long total = ((long long)E + Nn) * HEADS;
  if (tid >= total) return;
  const int h = (int)(tid & (HEADS - 1));
  const long long e = tid >> 3;
  int s, d;
  if (e < E) { s = ei[e]; d = ei[(size_t)E + e]; }
  else       { s = d = (int)(e - E); }
  const float* xj = H + ((size_t)s * HEADS + h) * C;
  const float* xi = H + ((size_t)d * HEADS + h) * C;
  const float* al = attL + h * C;
  const float* ar = attR + h * C;
  float logit = 0.f, av = 0.f;
  #pragma unroll
  for (int c = 0; c < C; ++c) {
    float vj = xj[c], vi = xi[c];
    logit += vi * vj;
    av += vj * al[c] + vi * ar[c];
  }
  float sig = 1.f / (1.f + __expf(-logit));
  av *= sig;
  av = av > 0.f ? av : NEG_SLOPE * av;
  alpha[tid] = av;
  atomicMax(&nmax[(size_t)d * HEADS + h], f2key(av));
}

// ---------------------------------------------------------------------------
// Pass 2 (fused): ex = exp(alpha - segmax[dst]); segment-sum ex; accumulate
// unnormalized ex * x_j into outAcc[dst]. Normalization happens node-wise in
// finalize (removes an entire third edge pass).
// ---------------------------------------------------------------------------
template <int C>
__global__ void edge_accum_kernel(const float* __restrict__ H,
                                  const int* __restrict__ ei,
                                  const float* __restrict__ alpha,
                                  const unsigned int* __restrict__ nmax,
                                  float* __restrict__ nsum,
                                  float* __restrict__ outAcc,
                                  int E, int Nn) {
  long long tid = (long long)blockIdx.x * blockDim.x + threadIdx.x;
  long long total = ((long long)E + Nn) * HEADS;
  if (tid >= total) return;
  const int h = (int)(tid & (HEADS - 1));
  const long long e = tid >> 3;
  int s, d;
  if (e < E) { s = ei[e]; d = ei[(size_t)E + e]; }
  else       { s = d = (int)(e - E); }
  float mx = key2f(nmax[(size_t)d * HEADS + h]);
  float ex = __expf(alpha[tid] - mx);
  atomicAdd(&nsum[(size_t)d * HEADS + h], ex);
  const float* xj = H + ((size_t)s * HEADS + h) * C;
  float* o = outAcc + ((size_t)d * HEADS + h) * C;
  #pragma unroll
  for (int c = 0; c < C; ++c) atomicAdd(&o[c], ex * xj[c]);
}

// Layer-1 finalize: normalize, +bias, ELU.
template <int C>
__global__ void finalize1_kernel(const float* __restrict__ outAcc,
                                 const float* __restrict__ nsum,
                                 const float* __restrict__ bias,
                                 float* __restrict__ Hout, int Nn) {
  const int F = HEADS * C;
  long long tid = (long long)blockIdx.x * blockDim.x + threadIdx.x;
  if (tid >= (long long)Nn * F) return;
  const int f = (int)(tid % F);
  const long long n = tid / F;
  const int h = f / C;
  float denom = nsum[n * HEADS + h] + 1e-16f;
  float v = outAcc[tid] / denom + bias[f];
  Hout[tid] = v > 0.f ? v : (__expf(v) - 1.f);   // ELU
}

// Layer-2 finalize: normalize, head-mean, +bias, log_softmax; also writes the
// trailing att_loss scalar (0.0 in inference).
__global__ void finalize2_kernel(const float* __restrict__ outAcc,
                                 const float* __restrict__ nsum,
                                 const float* __restrict__ bias,
                                 float* __restrict__ out, int Nn) {
  int n = blockIdx.x * blockDim.x + threadIdx.x;
  if (n >= Nn) return;
  float logits[7];
  float mx = -3.4e38f;
  #pragma unroll
  for (int c = 0; c < 7; ++c) {
    float s = 0.f;
    #pragma unroll
    for (int h = 0; h < HEADS; ++h) {
      size_t nh = (size_t)n * HEADS + h;
      s += outAcc[nh * 7 + c] / (nsum[nh] + 1e-16f);
    }
    float v = s * (1.0f / HEADS) + bias[c];
    logits[c] = v;
    mx = fmaxf(mx, v);
  }
  float se = 0.f;
  #pragma unroll
  for (int c = 0; c < 7; ++c) se += __expf(logits[c] - mx);
  float lse = mx + __logf(se);
  #pragma unroll
  for (int c = 0; c < 7; ++c) out[(size_t)n * 7 + c] = logits[c] - lse;
  if (n == 0) out[(size_t)Nn * 7] = 0.0f;   // att_loss
}

extern "C" void kernel_launch(void* const* d_in, const int* in_sizes, int n_in,
                              void* d_out, int out_size, void* d_ws, size_t ws_size,
                              hipStream_t stream) {
  (void)n_in; (void)out_size; (void)ws_size;
  const float* x   = (const float*)d_in[0];
  const int*   ei  = (const int*)d_in[1];
  const float* W1  = (const float*)d_in[2];
  const float* al1 = (const float*)d_in[3];
  const float* ar1 = (const float*)d_in[4];
  const float* b1  = (const float*)d_in[5];
  const float* W2  = (const float*)d_in[6];
  const float* al2 = (const float*)d_in[7];
  const float* ar2 = (const float*)d_in[8];
  const float* b2  = (const float*)d_in[9];
  float* out = (float*)d_out;

  const int F_IN = 256;
  const int Nn = in_sizes[0] / F_IN;      // 100000
  const int E  = in_sizes[1] / 2;         // 3200000
  const long long E2 = (long long)E + Nn; // + self loops
  const int F1 = HEADS * 8;               // 64
  const int F2 = HEADS * 7;               // 56

  // Workspace layout (floats); all region sizes are multiples of 8 floats so
  // every pointer below stays 32-byte aligned (needed for v16h loads).
  float* Hlin   = (float*)d_ws;                            // Nn*64 (layer2 reuses Nn*56)
  float* Hout1  = Hlin  + (size_t)Nn * F1;                 // Nn*64
  float* alphaB = Hout1 + (size_t)Nn * F1;                 // E2*HEADS
  unsigned int* nmax = (unsigned int*)(alphaB + (size_t)E2 * HEADS); // Nn*HEADS
  float* nsum   = (float*)(nmax + (size_t)Nn * HEADS);     // Nn*HEADS
  float* outAcc = nsum + (size_t)Nn * HEADS;               // Nn*64
  _Float16* Wp1 = (_Float16*)(outAcc + (size_t)Nn * F1);   // (256/32)*4*32*16 f16 = 32KB
  _Float16* Wp2 = Wp1 + (size_t)(F_IN / 32) * 4 * 32 * 16; // (64/32)*4*32*16 f16 = 8KB

  const long long eThreads = E2 * HEADS;
  const int eBlocks = (int)((eThreads + 255) / 256);

  // Pack weights into WMMA B-fragment order (tiny; L2-resident afterwards).
  pack_w_kernel<<<((F_IN >> 5) * 128 + 127) / 128, 128, 0, stream>>>(W1, Wp1, F_IN, F1);
  pack_w_kernel<<<((F1 >> 5) * 128 + 127) / 128, 128, 0, stream>>>(W2, Wp2, F1, F2);

  // ---------------- Layer 1 (C=8, concat) ----------------
  gemm_f16_wmma<<<dim3(Nn / 16), dim3(128), 0, stream>>>(x, (const v16h*)Wp1, Hlin, Nn, F_IN, F1);
  hipMemsetAsync(nmax,   0, (size_t)Nn * HEADS * sizeof(unsigned int), stream);
  hipMemsetAsync(nsum,   0, (size_t)Nn * HEADS * sizeof(float), stream);
  hipMemsetAsync(outAcc, 0, (size_t)Nn * F1 * sizeof(float), stream);
  edge_alpha_kernel<8><<<eBlocks, 256, 0, stream>>>(Hlin, ei, al1, ar1, alphaB, nmax, E, Nn);
  edge_accum_kernel<8><<<eBlocks, 256, 0, stream>>>(Hlin, ei, alphaB, nmax, nsum, outAcc, E, Nn);
  {
    long long t = (long long)Nn * F1;
    finalize1_kernel<8><<<(int)((t + 255) / 256), 256, 0, stream>>>(outAcc, nsum, b1, Hout1, Nn);
  }

  // ---------------- Layer 2 (C=7, head-mean) ----------------
  gemm_f16_wmma<<<dim3(Nn / 16), dim3(128), 0, stream>>>(Hout1, (const v16h*)Wp2, Hlin, Nn, F1, F2);
  hipMemsetAsync(nmax,   0, (size_t)Nn * HEADS * sizeof(unsigned int), stream);
  hipMemsetAsync(nsum,   0, (size_t)Nn * HEADS * sizeof(float), stream);
  hipMemsetAsync(outAcc, 0, (size_t)Nn * F2 * sizeof(float), stream);
  edge_alpha_kernel<7><<<eBlocks, 256, 0, stream>>>(Hlin, ei, al2, ar2, alphaB, nmax, E, Nn);
  edge_accum_kernel<7><<<eBlocks, 256, 0, stream>>>(Hlin, ei, alphaB, nmax, nsum, outAcc, E, Nn);
  finalize2_kernel<<<(Nn + 255) / 256, 256, 0, stream>>>(outAcc, nsum, b2, out, Nn);
}
